// GraphTrajectoryEncoder_81561428951534
// MI455X (gfx1250) — matrix operand order repaired and checked by reading, not compiled, not run on hardware
//
#include <hip/hip_runtime.h>
#include <hip/hip_bf16.h>
#include <math.h>
#include <stdint.h>

// ---------------------------------------------------------------------------
// Problem constants (match reference)
// ---------------------------------------------------------------------------
#define DEPTH 6
#define B_    4
#define S_    1024
#define D_    512
#define H_    8
#define HD_   64
#define INNER 512
#define F_    2048
#define MTOT  (B_ * S_)          // 4096 rows in all activation GEMMs

typedef __attribute__((ext_vector_type(16))) __bf16 v16bf;
typedef __attribute__((ext_vector_type(8)))  __bf16 v8bf;
typedef __attribute__((ext_vector_type(8)))  float  v8f;
typedef __attribute__((ext_vector_type(4)))  unsigned int u32x4;
typedef __attribute__((ext_vector_type(8)))  int i32x8;
typedef __attribute__((ext_vector_type(4)))  int i32x4;

#if defined(__AMDGCN__) && __has_builtin(__builtin_amdgcn_tensor_load_to_lds) && \
    __has_builtin(__builtin_amdgcn_s_wait_tensorcnt)
#define USE_TDM 1
#else
#define USE_TDM 0
#endif

static __device__ __forceinline__ __bf16 to_bf(float f) { return (__bf16)f; }

static __device__ __forceinline__ v16bf cat16(v8bf lo, v8bf hi) {
  return __builtin_shufflevector(lo, hi, 0,1,2,3,4,5,6,7,8,9,10,11,12,13,14,15);
}

static __device__ __forceinline__ v8f wmma_bf16(v16bf a, v16bf b, v8f c) {
  return __builtin_amdgcn_wmma_f32_16x16x32_bf16(
      /*neg_a=*/false, a, /*neg_b=*/false, b,
      /*c_mod=*/(short)0, c, /*reuse_a=*/false, /*reuse_b=*/false);
}

#if USE_TDM
// ---------------------------------------------------------------------------
// TDM: DMA a 2D bf16 tile [rows x tileK] (row stride = strideElems) from
// global memory into LDS at lds_off. D# packed per CDNA5 ISA ch.8.
// Issue from ONE wave per block; completion via TENSORcnt + block barrier.
// This toolchain exposes the 6-arg builtin:
//   (u32x4 g0, i32x8 g1, i32x4 g2, i32x4 g3, i32x8 extra, i32 cpol)
// ---------------------------------------------------------------------------
static __device__ __forceinline__ void tdm_load_tile_bf16(
    const __bf16* gptr, unsigned lds_off, unsigned tileK, unsigned rows,
    unsigned strideElems) {
  unsigned long long ga = (unsigned long long)(uintptr_t)gptr;
  u32x4 g0;
  g0[0] = 1u;                                   // count=1 (valid), user mode
  g0[1] = lds_off;                              // lds_addr (bytes)
  g0[2] = (unsigned)(ga & 0xffffffffu);         // global_addr[31:0]
  g0[3] = (unsigned)((ga >> 32) & 0x1ffffffu)   // global_addr[56:32]
          | (2u << 30);                         // type = 2 ("image")

  const unsigned td0 = tileK;                   // tensor_dim0 (covers tile)
  const unsigned td1 = rows;                    // tensor_dim1
  i32x8 g1;
  g1[0] = (int)(1u << 16);                      // data_size=1 -> 2 bytes
  g1[1] = (int)((td0 & 0xffffu) << 16);         // tensor_dim0[15:0] @ bit48
  g1[2] = (int)(((td0 >> 16) & 0xffffu) |       // tensor_dim0[31:16]
                ((td1 & 0xffffu) << 16));       // tensor_dim1[15:0]
  g1[3] = (int)(((td1 >> 16) & 0xffffu) |       // tensor_dim1[31:16]
                ((tileK & 0xffffu) << 16));     // tile_dim0
  g1[4] = (int)(rows & 0xffffu);                // tile_dim1 (tile_dim2=0)
  g1[5] = (int)strideElems;                     // tensor_dim0_stride[31:0]
  g1[6] = 0;                                    // stride[47:32], dim1_stride lo
  g1[7] = 0;

  i32x4 gz4 = {0, 0, 0, 0};                     // groups 2/3 unused (2D)
  i32x8 gz8 = {0, 0, 0, 0, 0, 0, 0, 0};
  __builtin_amdgcn_tensor_load_to_lds(g0, g1, gz4, gz4, gz8, 0);
}
#endif

// ---------------------------------------------------------------------------
// Weight conversion: W fp32 [DEPTH, K, N] -> Wt bf16 [DEPTH, N, K] (transpose)
// so WMMA B-operand / TDM tile loads are contiguous along K.
// ---------------------------------------------------------------------------
__global__ __launch_bounds__(256)
void convert_wt_kernel(const float* __restrict__ W, __bf16* __restrict__ Wt,
                       int K, int N) {
  const int dep = blockIdx.y;
  const size_t total = (size_t)K * N;
  size_t idx = (size_t)blockIdx.x * 256 + threadIdx.x;
  if (idx >= total) return;
  int n = (int)(idx % N);
  int k = (int)(idx / N);
  Wt[(size_t)dep * total + (size_t)n * K + k] =
      to_bf(W[(size_t)dep * total + idx]);
}

// ---------------------------------------------------------------------------
// LayerNorm: x fp32 [MTOT, 512] -> out bf16 [MTOT, 512]; one block per row.
// ---------------------------------------------------------------------------
__global__ __launch_bounds__(256)
void layernorm_kernel(const float* __restrict__ x, const float* __restrict__ w,
                      const float* __restrict__ bsh, __bf16* __restrict__ out) {
  const int row = blockIdx.x;
  const int tid = threadIdx.x;
  const float* xr = x + (size_t)row * D_;
  float a0 = xr[tid];
  float a1 = xr[tid + 256];

  __shared__ float red[256];
  red[tid] = a0 + a1;
  __syncthreads();
  for (int off = 128; off > 0; off >>= 1) {
    if (tid < off) red[tid] += red[tid + off];
    __syncthreads();
  }
  float mu = red[0] * (1.0f / D_);
  __syncthreads();

  float d0 = a0 - mu, d1 = a1 - mu;
  red[tid] = d0 * d0 + d1 * d1;
  __syncthreads();
  for (int off = 128; off > 0; off >>= 1) {
    if (tid < off) red[tid] += red[tid + off];
    __syncthreads();
  }
  float rstd = rsqrtf(red[0] * (1.0f / D_) + 1e-5f);

  __bf16* orow = out + (size_t)row * D_;
  orow[tid]       = to_bf(d0 * rstd * w[tid]       + bsh[tid]);
  orow[tid + 256] = to_bf(d1 * rstd * w[tid + 256] + bsh[tid + 256]);
}

// ---------------------------------------------------------------------------
// WMMA GEMM:  C[M,N] = A[M,K](bf16) @ Wt[N,K](bf16, pre-transposed) (+bias)
// MODE 0: bf16 out (+bias if bias!=null)
// MODE 1: bf16 out, exact GELU(acc+bias)
// MODE 2: fp32 out = acc + bias + res   (residual add)
// Block = 256 threads = 8 waves stacked in M; wave tile = 16x64.
// Weight tile [64N x 64K] staged in LDS by the Tensor Data Mover,
// double-buffered, shared by all 8 waves. grid = (N/64, M/128).
// ---------------------------------------------------------------------------
template <int MODE>
__global__ __launch_bounds__(256)
void gemm_wmma_kernel(const __bf16* __restrict__ A, const __bf16* __restrict__ Wt,
                      const float* __restrict__ bias, const float* __restrict__ res,
                      void* __restrict__ out, int M, int N, int K) {
  const int lane = threadIdx.x & 31;
  const int wave = threadIdx.x >> 5;
  const int lm   = lane & 15;
  const int hi   = lane >> 4;               // 0 or 1
  const int mbase = blockIdx.y * 128 + wave * 16;
  const int nbase = blockIdx.x * 64;

  v8f acc[4];
#pragma unroll
  for (int c = 0; c < 4; ++c) acc[c] = v8f{0.f,0.f,0.f,0.f,0.f,0.f,0.f,0.f};

  const __bf16* arow = A + (size_t)(mbase + lm) * K + hi * 16;

#if USE_TDM
  __shared__ __bf16 bsm[2][64][64];         // 2 x 8KB weight tiles
  const __bf16* wtile0 = Wt + (size_t)nbase * K;
  const int nk = K >> 6;                    // 64-wide K super-steps

  if (wave == 0)
    tdm_load_tile_bf16(wtile0, (unsigned)(uintptr_t)&bsm[0][0][0],
                       64u, 64u, (unsigned)K);

  for (int i = 0; i < nk; ++i) {
    if (wave == 0) {
      if (i + 1 < nk) {
        tdm_load_tile_bf16(wtile0 + (size_t)(i + 1) * 64,
                           (unsigned)(uintptr_t)&bsm[(i + 1) & 1][0][0],
                           64u, 64u, (unsigned)K);
        __builtin_amdgcn_s_wait_tensorcnt(1);   // tile i resident (in-order)
      } else {
        __builtin_amdgcn_s_wait_tensorcnt(0);
      }
    }
    __syncthreads();                         // tile i visible to all waves

    const int kb = i * 64;
#pragma unroll
    for (int kk = 0; kk < 64; kk += 32) {
      v8bf alo = *(const v8bf*)(arow + kb + kk);
      v8bf ahi = *(const v8bf*)(arow + kb + kk + 8);
      v16bf a = cat16(alo, ahi);
#pragma unroll
      for (int c = 0; c < 4; ++c) {
        const __bf16* wrow = &bsm[i & 1][c * 16 + lm][kk + hi * 16];
        v8bf b0 = *(const v8bf*)(wrow);
        v8bf b1 = *(const v8bf*)(wrow + 8);
        acc[c] = wmma_bf16(a, cat16(b0, b1), acc[c]);
      }
    }
    __syncthreads();                         // done reading buf before reuse
  }
#else
  for (int kb = 0; kb < K; kb += 32) {
    v8bf alo = *(const v8bf*)(arow + kb);
    v8bf ahi = *(const v8bf*)(arow + kb + 8);
    v16bf a = cat16(alo, ahi);
#pragma unroll
    for (int c = 0; c < 4; ++c) {
      const __bf16* wrow = Wt + (size_t)(nbase + c * 16 + lm) * K + kb + hi * 16;
      v8bf b0 = *(const v8bf*)(wrow);
      v8bf b1 = *(const v8bf*)(wrow + 8);
      acc[c] = wmma_bf16(a, cat16(b0, b1), acc[c]);
    }
  }
#endif

#pragma unroll
  for (int c = 0; c < 4; ++c) {
    const int n = nbase + c * 16 + lm;
    const float bv = bias ? bias[n] : 0.0f;
#pragma unroll
    for (int r = 0; r < 8; ++r) {
      const int m = mbase + r + hi * 8;
      const size_t off = (size_t)m * N + n;
      float v = acc[c][r] + bv;
      if (MODE == 0) {
        ((__bf16*)out)[off] = to_bf(v);
      } else if (MODE == 1) {
        float g = 0.5f * v * (1.0f + erff(v * 0.70710678118654752f));
        ((__bf16*)out)[off] = to_bf(g);
      } else {
        ((float*)out)[off] = v + res[off];
      }
    }
  }
}

// ---------------------------------------------------------------------------
// Transpose V slice of qkv into vt[b][h][d][t] (bf16) for contiguous PV loads.
// ---------------------------------------------------------------------------
__global__ __launch_bounds__(256)
void transpose_v_kernel(const __bf16* __restrict__ qkv, __bf16* __restrict__ vt) {
  size_t idx = (size_t)blockIdx.x * 256 + threadIdx.x;   // B*H*HD*S = 2^21
  int t  = (int)(idx & (S_ - 1));
  int dd = (int)((idx >> 10) & (HD_ - 1));
  int h  = (int)((idx >> 16) & (H_ - 1));
  int b  = (int)(idx >> 19);
  vt[idx] = qkv[((size_t)(b * S_ + t)) * (3 * INNER) + 2 * INNER + h * HD_ + dd];
}

// ---------------------------------------------------------------------------
// Flash-style masked attention.
// grid = (S/64, H, B), block = 128 (4 waves); each wave owns a 16-row Q tile.
// qkv bf16 [B*S, 1536] (q @ 0, k @ 512, v @ 1024), vt bf16 [B,H,64,1024].
// o bf16 [B*S, 512] (head-major inner dim).
// ---------------------------------------------------------------------------
__global__ __launch_bounds__(128)
void attn_kernel(const __bf16* __restrict__ qkv, const __bf16* __restrict__ vt,
                 const int* __restrict__ gmask, __bf16* __restrict__ o) {
  const int lane = threadIdx.x & 31;
  const int wave = threadIdx.x >> 5;
  const int lm   = lane & 15;
  const int hi   = lane >> 4;
  const int b    = blockIdx.z;
  const int hh   = blockIdx.y;
  const int qbase = (blockIdx.x * 4 + wave) * 16;
  const int hoff  = hh * HD_;
  const float scale = 0.125f;                // HD^-0.5

  __shared__ __bf16 pbuf[4][16][32];         // per-wave probs transpose buffer

  // Q A-tiles (16x32 each) covering d 0..31 and 32..63
  const __bf16* qrow = qkv + (size_t)(b * S_ + qbase + lm) * (3 * INNER) + hoff;
  v16bf qa[2];
#pragma unroll
  for (int p = 0; p < 2; ++p) {
    v8bf lo = *(const v8bf*)(qrow + p * 32 + hi * 16);
    v8bf hi8 = *(const v8bf*)(qrow + p * 32 + 8 + hi * 16);
    qa[p] = cat16(lo, hi8);
  }

  v8f acc[4];
#pragma unroll
  for (int c = 0; c < 4; ++c) acc[c] = v8f{0.f,0.f,0.f,0.f,0.f,0.f,0.f,0.f};
  float mrow[8], lsum[8];
#pragma unroll
  for (int r = 0; r < 8; ++r) { mrow[r] = -INFINITY; lsum[r] = 0.0f; }

  const __bf16* kbase = qkv + (size_t)b * S_ * (3 * INNER) + INNER + hoff;
  const __bf16* vtb   = vt + (size_t)(b * H_ + hh) * HD_ * S_;

  for (int kt0 = 0; kt0 < S_; kt0 += 32) {
    // --- scores: two 16x16 C tiles covering kt columns [kt0,kt0+16) and [+16,+32)
    v8f s[2];
#pragma unroll
    for (int j = 0; j < 2; ++j) {
      s[j] = v8f{0.f,0.f,0.f,0.f,0.f,0.f,0.f,0.f};
      const int t = kt0 + j * 16 + lm;
      const __bf16* kr = kbase + (size_t)t * (3 * INNER) + hi * 16;
#pragma unroll
      for (int p = 0; p < 2; ++p) {
        v8bf b0 = *(const v8bf*)(kr + p * 32);
        v8bf b1 = *(const v8bf*)(kr + p * 32 + 8);
        s[j] = wmma_bf16(qa[p], cat16(b0, b1), s[j]);
      }
    }

    // --- mask + online softmax (row stats via 16-lane shuffles)
#pragma unroll
    for (int r = 0; r < 8; ++r) {
      const int q  = qbase + r + hi * 8;
      const int t0 = kt0 + lm;
      const int t1 = t0 + 16;
      const int* gr = gmask + ((size_t)b * S_ + q) * S_ + kt0;
      float v0 = s[0][r] * scale;
      float v1 = s[1][r] * scale;
      v0 = ((gr[lm] > 0)      || (q == t0)) ? v0 : -1e9f;
      v1 = ((gr[16 + lm] > 0) || (q == t1)) ? v1 : -1e9f;

      float tmax = fmaxf(v0, v1);
      tmax = fmaxf(tmax, __shfl_xor(tmax, 1));
      tmax = fmaxf(tmax, __shfl_xor(tmax, 2));
      tmax = fmaxf(tmax, __shfl_xor(tmax, 4));
      tmax = fmaxf(tmax, __shfl_xor(tmax, 8));

      float mnew  = fmaxf(mrow[r], tmax);
      float alpha = __expf(mrow[r] - mnew);
      float p0 = __expf(v0 - mnew);
      float p1 = __expf(v1 - mnew);
      float tsum = p0 + p1;
      tsum += __shfl_xor(tsum, 1);
      tsum += __shfl_xor(tsum, 2);
      tsum += __shfl_xor(tsum, 4);
      tsum += __shfl_xor(tsum, 8);

      lsum[r] = lsum[r] * alpha + tsum;
      mrow[r] = mnew;
#pragma unroll
      for (int c = 0; c < 4; ++c) acc[c][r] *= alpha;

      const int rowi = r + hi * 8;
      pbuf[wave][rowi][lm]      = to_bf(p0);
      pbuf[wave][rowi][16 + lm] = to_bf(p1);
    }
    __syncthreads();   // make LDS stores visible before A-layout reload

    // --- probs C-layout -> A-layout via LDS
    const __bf16* pr = &pbuf[wave][lm][hi * 16];
    v8bf pl0 = *(const v8bf*)(pr);
    v8bf pl1 = *(const v8bf*)(pr + 8);
    v16bf pa = cat16(pl0, pl1);

    // --- PV: 4 C tiles across d chunks of 16
#pragma unroll
    for (int c = 0; c < 4; ++c) {
      const __bf16* vr = vtb + (size_t)(c * 16 + lm) * S_ + kt0 + hi * 16;
      v8bf b0 = *(const v8bf*)(vr);
      v8bf b1 = *(const v8bf*)(vr + 8);
      acc[c] = wmma_bf16(pa, cat16(b0, b1), acc[c]);
    }
    __syncthreads();
  }

  // --- finalize: divide by row sums, store o[b,s, h*64 + d]
#pragma unroll
  for (int r = 0; r < 8; ++r) {
    const float inv = 1.0f / lsum[r];
    const int q = qbase + r + hi * 8;
    __bf16* orow = o + (size_t)(b * S_ + q) * INNER + hoff;
#pragma unroll
    for (int c = 0; c < 4; ++c)
      orow[c * 16 + lm] = to_bf(acc[c][r] * inv);
  }
}

// ---------------------------------------------------------------------------
// Host-side orchestration
// ---------------------------------------------------------------------------
extern "C" void kernel_launch(void* const* d_in, const int* in_sizes, int n_in,
                              void* d_out, int out_size, void* d_ws, size_t ws_size,
                              hipStream_t stream) {
  (void)in_sizes; (void)n_in; (void)out_size; (void)ws_size;

  const float* x     = (const float*)d_in[0];
  const int*   gmask = (const int*)  d_in[1];
  const float* ln1w  = (const float*)d_in[2];
  const float* ln1b  = (const float*)d_in[3];
  const float* Wqkv  = (const float*)d_in[4];
  const float* Wout  = (const float*)d_in[5];
  const float* bout  = (const float*)d_in[6];
  const float* ln2w  = (const float*)d_in[7];
  const float* ln2b  = (const float*)d_in[8];
  const float* W1    = (const float*)d_in[9];
  const float* b1    = (const float*)d_in[10];
  const float* W2    = (const float*)d_in[11];
  const float* b2    = (const float*)d_in[12];

  char* ws = (char*)d_ws;
  size_t off = 0;
  auto carve = [&](size_t bytes) -> char* {
    char* p = ws + off;
    off += (bytes + 255) & ~(size_t)255;
    return p;
  };

  const size_t QKV_W = (size_t)1536 * 512;   // per-layer Wqkv_t elems
  const size_t OUT_W = (size_t)512 * 512;
  const size_t W1_W  = (size_t)2048 * 512;
  const size_t W2_W  = (size_t)512 * 2048;

  __bf16* wqkv_t = (__bf16*)carve(DEPTH * QKV_W * 2);
  __bf16* wout_t = (__bf16*)carve(DEPTH * OUT_W * 2);
  __bf16* w1_t   = (__bf16*)carve(DEPTH * W1_W * 2);
  __bf16* w2_t   = (__bf16*)carve(DEPTH * W2_W * 2);
  float*  hbuf   = (float*) carve((size_t)MTOT * D_ * 4);
  __bf16* nrm    = (__bf16*)carve((size_t)MTOT * D_ * 2);
  __bf16* qkvb   = (__bf16*)carve((size_t)MTOT * 3 * INNER * 2);
  __bf16* vtb    = (__bf16*)carve((size_t)B_ * H_ * HD_ * S_ * 2);
  __bf16* obuf   = (__bf16*)carve((size_t)MTOT * INNER * 2);
  __bf16* ffb    = (__bf16*)carve((size_t)MTOT * F_ * 2);

  // Weight conversion (transpose + fp32->bf16), all layers per launch.
  convert_wt_kernel<<<dim3((512 * 1536 + 255) / 256, DEPTH), 256, 0, stream>>>(Wqkv, wqkv_t, 512, 1536);
  convert_wt_kernel<<<dim3((512 * 512  + 255) / 256, DEPTH), 256, 0, stream>>>(Wout, wout_t, 512, 512);
  convert_wt_kernel<<<dim3((512 * 2048 + 255) / 256, DEPTH), 256, 0, stream>>>(W1,   w1_t,   512, 2048);
  convert_wt_kernel<<<dim3((2048 * 512 + 255) / 256, DEPTH), 256, 0, stream>>>(W2,   w2_t,   2048, 512);

  for (int d = 0; d < DEPTH; ++d) {
    const float* hsrc = (d == 0) ? x : hbuf;

    // --- attention block ---
    layernorm_kernel<<<MTOT, 256, 0, stream>>>(hsrc, ln1w + d * D_, ln1b + d * D_, nrm);

    gemm_wmma_kernel<0><<<dim3(1536 / 64, MTOT / 128), 256, 0, stream>>>(
        nrm, wqkv_t + (size_t)d * QKV_W, nullptr, nullptr, (void*)qkvb,
        MTOT, 1536, 512);

    transpose_v_kernel<<<(B_ * H_ * HD_ * S_) / 256, 256, 0, stream>>>(qkvb, vtb);

    attn_kernel<<<dim3(S_ / 64, H_, B_), 128, 0, stream>>>(qkvb, vtb, gmask, obuf);

    gemm_wmma_kernel<2><<<dim3(512 / 64, MTOT / 128), 256, 0, stream>>>(
        obuf, wout_t + (size_t)d * OUT_W, bout + d * D_, hsrc, (void*)hbuf,
        MTOT, 512, 512);

    // --- feedforward block ---
    layernorm_kernel<<<MTOT, 256, 0, stream>>>(hbuf, ln2w + d * D_, ln2b + d * D_, nrm);

    gemm_wmma_kernel<1><<<dim3(2048 / 64, MTOT / 128), 256, 0, stream>>>(
        nrm, w1_t + (size_t)d * W1_W, b1 + d * F_, nullptr, (void*)ffb,
        MTOT, 2048, 512);

    void* dst = (d == DEPTH - 1) ? d_out : (void*)hbuf;
    gemm_wmma_kernel<2><<<dim3(512 / 64, MTOT / 128), 256, 0, stream>>>(
        ffb, w2_t + (size_t)d * W2_W, b2 + d * D_, hbuf, dst,
        MTOT, 512, 2048);
  }
}
